// Decoder_26130581029031
// MI455X (gfx1250) — matrix operand order repaired and checked by reading, not compiled, not run on hardware
//
#include <hip/hip_runtime.h>

// ---------------------------------------------------------------------------
// Types for CDNA5 WMMA (wave32): v_wmma_f32_16x16x32_bf16
// ---------------------------------------------------------------------------
typedef __attribute__((ext_vector_type(16))) __bf16 v16bf;
typedef __attribute__((ext_vector_type(4)))  __bf16 v4bf;
typedef __attribute__((ext_vector_type(8)))  float  v8f;

#define DEV __device__ __forceinline__

static constexpr int B_    = 64;
static constexpr int T_    = 800;
static constexpr int IN_   = 128;
static constexpr int H_    = 256;
static constexpr int MELN  = 400;   // FRAME*R
static constexpr int STOPN = 656;   // H + MEL

DEV float sigmoidf_(float x){ return 1.f / (1.f + __expf(-x)); }

// ---------------------------------------------------------------------------
// Fragment index math (CDNA5 ISA 7.12.2, 16-bit operands, wave32)
// A (16xK, M=lane&15):   K = (e&7) | (e>>3)<<4 | (lane>>4)<<3
// B (Kx16, N=lane&15):   K = (e&7) | (e>>3)<<4 | (lane>>4)<<3 (same as A)
// Fragment block = 512 bf16 (1 KB), lane-major: [lane][e]
// ---------------------------------------------------------------------------
DEV void stageA(__bf16* frag, int m, int k, float v){
  int kt = k >> 5, kk = k & 31;
  int lane = (m & 15) | (((kk >> 3) & 1) << 4);
  int e    = (kk & 7) | (((kk >> 4) & 1) << 3);
  frag[kt * 512 + lane * 16 + e] = (__bf16)v;
}

DEV void stage_from(__bf16* frag, const float* src, int K, int tid){
  for (int u = tid; u < 16 * K; u += 512){
    int m = u / K, k = u - m * K;
    stageA(frag, m, k, src[u]);
  }
}

// C = A(16xK) * W^T tile, W pre-packed in B-fragment order in global memory.
DEV void gemm_bf16(const __bf16* frag, const __bf16* __restrict__ Wp,
                   int nNT, int nKT, const float* __restrict__ bias,
                   float* out, int ldOut, int relu, int wave, int lane){
  for (int nt = wave; nt < nNT; nt += 16){
    v8f c = {};
    const __bf16* bp = Wp + (size_t)nt * nKT * 512 + lane * 16;
    const __bf16* ap = frag + lane * 16;
    for (int kt = 0; kt < nKT; ++kt){
      v16bf a = *(const v16bf*)(ap + kt * 512);
      v16bf b = *(const v16bf*)(bp + kt * 512);
      c = __builtin_amdgcn_wmma_f32_16x16x32_bf16(false, a, false, b,
                                                  (short)0, c, false, false);
    }
    int n  = nt * 16 + (lane & 15);
    int mb = (lane >> 4) * 8;
    float bv = bias[n];
#pragma unroll
    for (int r = 0; r < 8; ++r){
      float v = c[r] + bv;
      if (relu) v = fmaxf(v, 0.f);
      out[(mb + r) * ldOut + n] = v;
    }
  }
}

// GRU gate pointwise. mode 0: plain (stage h'); 1: dec+=h', stage dec;
// 2: decOut=dec+h', stage decOut.
DEV void gru_pointwise(const float* gi, const float* gh, float* h,
                       float* dec, float* decOut, __bf16* frag,
                       int tid, int mode){
  for (int u = tid; u < 16 * H_; u += 512){
    int m = u >> 8, j = u & 255;
    const float* gim = gi + m * 768;
    const float* ghm = gh + m * 768;
    float r  = sigmoidf_(gim[j]       + ghm[j]);
    float z  = sigmoidf_(gim[256 + j] + ghm[256 + j]);
    float nn = tanhf   (gim[512 + j] + r * ghm[512 + j]);
    float hn = (1.f - z) * nn + z * h[u];
    h[u] = hn;
    float sv = hn;
    if (mode == 1){ sv = dec[u] + hn; dec[u] = sv; }
    else if (mode == 2){ sv = dec[u] + hn; decOut[u] = sv; }
    stageA(frag, m, j, sv);
  }
}

// ---------------------------------------------------------------------------
// Phase 0: pack a row-major weight W[N][K] (fp32) into bf16 B-fragment order.
// ---------------------------------------------------------------------------
__global__ void pack_w(const float* __restrict__ W, __bf16* __restrict__ dst,
                       int N, int K, int Kpad){
  int total = N * Kpad;
  int nKT = Kpad >> 5;
  for (int u = blockIdx.x * blockDim.x + threadIdx.x; u < total;
       u += gridDim.x * blockDim.x){
    int n = u / Kpad, k = u - n * Kpad;
    float v = (k < K) ? W[(size_t)n * K + k] : 0.f;
    int nt = n >> 4, kt = k >> 5, kk = k & 31;
    int lane = (n & 15) | (((kk >> 3) & 1) << 4);
    int e    = (kk & 7) | (((kk >> 4) & 1) << 3);
    dst[((size_t)(nt * nKT + kt)) * 512 + lane * 16 + e] = (__bf16)v;
  }
}

__global__ void cvt_bf16(const float* __restrict__ src,
                         __bf16* __restrict__ dst, int n){
  for (int i = blockIdx.x * blockDim.x + threadIdx.x; i < n;
       i += gridDim.x * blockDim.x)
    dst[i] = (__bf16)src[i];
}

// ---------------------------------------------------------------------------
// Phase 1: proc_inputs = x @ att_Wi^T + att_bi   -> bf16 [B*T][128]
// 8 waves/block, one 16-row M-tile per wave; WMMA over K=128 (4 K-tiles).
// ---------------------------------------------------------------------------
__global__ __launch_bounds__(256) void proc_kernel(
    const float* __restrict__ x, const __bf16* __restrict__ wi_p,
    const float* __restrict__ att_bi, __bf16* __restrict__ procbf){
  __shared__ __align__(64) __bf16 sA[8 * 2048];
  int tid = threadIdx.x, wave = tid >> 5, lane = tid & 31;
  int row0 = (blockIdx.x * 8 + wave) * 16;
  __bf16* frag = sA + wave * 2048;
  for (int u = lane; u < 16 * 128; u += 32){
    int m = u >> 7, k = u & 127;
    stageA(frag, m, k, x[(size_t)(row0 + m) * 128 + k]);
  }
  __syncthreads();
  for (int nt = 0; nt < 8; ++nt){
    v8f c = {};
    const __bf16* bp = wi_p + (size_t)nt * 4 * 512 + lane * 16;
    for (int kt = 0; kt < 4; ++kt){
      v16bf a = *(const v16bf*)(frag + kt * 512 + lane * 16);
      v16bf b = *(const v16bf*)(bp + kt * 512);
      c = __builtin_amdgcn_wmma_f32_16x16x32_bf16(false, a, false, b,
                                                  (short)0, c, false, false);
    }
    int n = nt * 16 + (lane & 15), mb = (lane >> 4) * 8;
    float bv = att_bi[n];
#pragma unroll
    for (int r = 0; r < 8; ++r)
      procbf[(size_t)(row0 + mb + r) * 128 + n] = (__bf16)(c[r] + bv);
  }
}

// ---------------------------------------------------------------------------
// Phase 2: persistent decoder. 4 blocks x 512 threads (16 waves), each block
// owns a 16-row batch tile and runs all 800 steps; all state in LDS.
// ---------------------------------------------------------------------------
__global__ __launch_bounds__(512) void decoder_kernel(
    const __bf16* __restrict__ xbf, const __bf16* __restrict__ procbf,
    const __bf16* pre1_p, const float* pre_b1,
    const __bf16* pre2_p, const float* pre_b2,
    const __bf16* aWih_p, const __bf16* aWhh_p,
    const float* a_bih, const float* a_bhh,
    const __bf16* wq_p, const float* att_bq, const float* att_v,
    const __bf16* proj_p, const float* proj_b,
    const __bf16* d1Wih_p, const __bf16* d1Whh_p,
    const float* d1_bih, const float* d1_bhh,
    const __bf16* d2Wih_p, const __bf16* d2Whh_p,
    const float* d2_bih, const float* d2_bhh,
    const __bf16* mel_p, const float* mel_b,
    const float* stop_W, const float* stop_b,
    float* __restrict__ outp, float* __restrict__ stops){

  __shared__ float sHa[4096], sHd1[4096], sHd2[4096];   // GRU hidden states
  __shared__ float sFrame[1536];                        // frame, K padded to 96
  __shared__ __align__(64) __bf16 sAfrag[4096];         // A operand staging
  __shared__ float sRegA[12800];                        // gi / scores / mel out
  __shared__ float sRegB[12288];                        // gh / align(bf16)
  __shared__ float sU[4096];                            // pre / q+ctx / dec_out
  __shared__ float sDec[4096];                          // dec_in
  __shared__ float sRed[544];                           // reductions
  __shared__ float sVv[128];                            // att_v

  const int tid = threadIdx.x, wave = tid >> 5, lane = tid & 31;
  const int b0 = blockIdx.x * 16;

  for (int u = tid; u < 4096; u += 512){ sHa[u]=0.f; sHd1[u]=0.f; sHd2[u]=0.f; sDec[u]=0.f; }
  for (int u = tid; u < 1536; u += 512) sFrame[u] = 0.f;
  if (tid < 128) sVv[tid] = att_v[tid];
  __syncthreads();

  for (int t = 0; t < T_; ++t){
    // ---------------- prenet ----------------
    stage_from(sAfrag, sFrame, 96, tid);                              __syncthreads();
    gemm_bf16(sAfrag, pre1_p, 16, 3, pre_b1, sU, 256, 1, wave, lane); __syncthreads();
    stage_from(sAfrag, sU, 256, tid);                                 __syncthreads();
    gemm_bf16(sAfrag, pre2_p, 8, 8, pre_b2, sU, 128, 1, wave, lane);  __syncthreads();
    stage_from(sAfrag, sU, 128, tid);                                 __syncthreads();

    // ---------------- attention GRU ----------------
    gemm_bf16(sAfrag, aWih_p, 48, 4, a_bih, sRegA, 768, 0, wave, lane); __syncthreads();
    stage_from(sAfrag, sHa, 256, tid);                                  __syncthreads();
    gemm_bf16(sAfrag, aWhh_p, 48, 8, a_bhh, sRegB, 768, 0, wave, lane); __syncthreads();
    gru_pointwise(sRegA, sRegB, sHa, nullptr, nullptr, sAfrag, tid, 0); __syncthreads();

    // ---------------- query ----------------
    gemm_bf16(sAfrag, wq_p, 8, 8, att_bq, sU, 128, 0, wave, lane);      __syncthreads();

    // ---------------- attention scores (per-row, VALU tanh) ----------------
    {
      int m = lane & 15, half = lane >> 4;
      const float* pq = sU + m * 128 + half * 64;
      const float* vv = sVv + half * 64;
      for (int tt = 0; tt < 50; ++tt){
        int tpos = wave * 50 + tt;
        const __bf16* pr = procbf + ((size_t)(b0 + m) * T_ + tpos) * 128 + half * 64;
        float acc = 0.f;
        for (int aa = 0; aa < 64; aa += 4){
          v4bf p = *(const v4bf*)(pr + aa);
          acc += vv[aa+0] * tanhf(pq[aa+0] + (float)p[0]);
          acc += vv[aa+1] * tanhf(pq[aa+1] + (float)p[1]);
          acc += vv[aa+2] * tanhf(pq[aa+2] + (float)p[2]);
          acc += vv[aa+3] * tanhf(pq[aa+3] + (float)p[3]);
        }
        acc += __shfl_xor(acc, 16);
        if (half == 0) sRegA[m * T_ + tpos] = acc;
      }
    }
    __syncthreads();

    // ---------------- softmax over T ----------------
    {
      int m = tid & 15, ch = tid >> 4;                  // 32 chunks of 25
      float mx = -3.0e38f;
      for (int q = 0; q < 25; ++q) mx = fmaxf(mx, sRegA[m * T_ + ch * 25 + q]);
      sRed[ch * 16 + m] = mx; __syncthreads();
      if (tid < 16){
        float v = -3.0e38f;
        for (int c2 = 0; c2 < 32; ++c2) v = fmaxf(v, sRed[c2 * 16 + tid]);
        sRed[512 + tid] = v;
      }
      __syncthreads();
      float rmax = sRed[512 + m];
      float sm = 0.f;
      for (int q = 0; q < 25; ++q) sm += __expf(sRegA[m * T_ + ch * 25 + q] - rmax);
      sRed[ch * 16 + m] = sm; __syncthreads();
      if (tid < 16){
        float v = 0.f;
        for (int c2 = 0; c2 < 32; ++c2) v += sRed[c2 * 16 + tid];
        sRed[528 + tid] = v;
      }
      __syncthreads();
      float inv = 1.f / sRed[528 + m];
      __bf16* al = (__bf16*)sRegB;
      for (int q = 0; q < 25; ++q){
        int tp = ch * 25 + q;
        al[m * T_ + tp] = (__bf16)(__expf(sRegA[m * T_ + tp] - rmax) * inv);
      }
    }
    __syncthreads();

    // ---------------- context: per-row matvec over T (bf16 x from L2) -------
    {
      const __bf16* al = (const __bf16*)sRegB + wave * T_;
      const __bf16* xr = xbf + (size_t)(b0 + wave) * T_ * IN_ + lane * 4;
      float a0=0.f, a1=0.f, a2=0.f, a3=0.f;
      for (int tp = 0; tp < T_; ++tp){
        float w = (float)al[tp];
        v4bf xv = *(const v4bf*)(xr + (size_t)tp * IN_);
        a0 += w * (float)xv[0]; a1 += w * (float)xv[1];
        a2 += w * (float)xv[2]; a3 += w * (float)xv[3];
      }
      float* ctx = sU + 2048 + wave * 128 + lane * 4;
      ctx[0] = a0; ctx[1] = a1; ctx[2] = a2; ctx[3] = a3;
    }
    __syncthreads();

    // ---------------- projection -> dec_in ----------------
    stage_from(sAfrag, sU + 2048, 128, tid);                             __syncthreads();
    gemm_bf16(sAfrag, proj_p, 16, 4, proj_b, sDec, 256, 0, wave, lane);  __syncthreads();
    stage_from(sAfrag, sDec, 256, tid);                                  __syncthreads();

    // ---------------- decoder GRU 1 ----------------
    gemm_bf16(sAfrag, d1Wih_p, 48, 8, d1_bih, sRegA, 768, 0, wave, lane); __syncthreads();
    stage_from(sAfrag, sHd1, 256, tid);                                   __syncthreads();
    gemm_bf16(sAfrag, d1Whh_p, 48, 8, d1_bhh, sRegB, 768, 0, wave, lane); __syncthreads();
    gru_pointwise(sRegA, sRegB, sHd1, sDec, nullptr, sAfrag, tid, 1);     __syncthreads();

    // ---------------- decoder GRU 2 ----------------
    gemm_bf16(sAfrag, d2Wih_p, 48, 8, d2_bih, sRegA, 768, 0, wave, lane); __syncthreads();
    stage_from(sAfrag, sHd2, 256, tid);                                   __syncthreads();
    gemm_bf16(sAfrag, d2Whh_p, 48, 8, d2_bhh, sRegB, 768, 0, wave, lane); __syncthreads();
    gru_pointwise(sRegA, sRegB, sHd2, sDec, sU, sAfrag, tid, 2);          __syncthreads();

    // ---------------- mel projection, fused transpose store ----------------
    for (int nt = wave; nt < 25; nt += 16){
      v8f c = {};
      const __bf16* bp = mel_p + (size_t)nt * 8 * 512 + lane * 16;
      const __bf16* ap = sAfrag + lane * 16;
      for (int kt = 0; kt < 8; ++kt){
        v16bf a = *(const v16bf*)(ap + kt * 512);
        v16bf b = *(const v16bf*)(bp + kt * 512);
        c = __builtin_amdgcn_wmma_f32_16x16x32_bf16(false, a, false, b,
                                                    (short)0, c, false, false);
      }
      int n = nt * 16 + (lane & 15), mb = (lane >> 4) * 8;
      float bv = mel_b[n];
      int r5 = n / 80, f = n - r5 * 80;
#pragma unroll
      for (int r = 0; r < 8; ++r){
        int m = mb + r;
        float v = c[r] + bv;
        sRegA[m * MELN + n] = v;
        // outputs[b][f][t*5 + r5] = out_step[b][r5*80 + f]
        outp[(size_t)(b0 + m) * 320000 + (size_t)f * 4000 + t * 5 + r5] = v;
      }
    }
    __syncthreads();

    // ---------------- stop token ----------------
    {
      int m = tid & 15, part = tid >> 4;
      float s = 0.f;
      int i0 = part * 21, i1 = (i0 + 21 < STOPN) ? i0 + 21 : STOPN;
      for (int i = i0; i < i1; ++i){
        float v = (i < H_) ? sU[m * H_ + i] : sRegA[m * MELN + (i - H_)];
        s += v * stop_W[i];
      }
      sRed[part * 16 + m] = s; __syncthreads();
      if (tid < 16){
        float tot = stop_b[0];
        for (int p2 = 0; p2 < 32; ++p2) tot += sRed[p2 * 16 + tid];
        stops[(size_t)(b0 + tid) * T_ + t] = sigmoidf_(tot);
      }
    }
    __syncthreads();

    // ---------------- next frame = out[:, 320:400], zero-padded to 96 ------
    for (int u = tid; u < 1536; u += 512){
      int m = u / 96, f = u - m * 96;
      sFrame[u] = (f < 80) ? sRegA[m * MELN + 320 + f] : 0.f;
    }
    __syncthreads();
  }
}

// ---------------------------------------------------------------------------
// Host launcher
// ---------------------------------------------------------------------------
extern "C" void kernel_launch(void* const* d_in, const int* in_sizes, int n_in,
                              void* d_out, int out_size, void* d_ws, size_t ws_size,
                              hipStream_t stream){
  const float* x       = (const float*)d_in[0];
  const float* pre_W1  = (const float*)d_in[1];
  const float* pre_b1  = (const float*)d_in[2];
  const float* pre_W2  = (const float*)d_in[3];
  const float* pre_b2  = (const float*)d_in[4];
  const float* aWih    = (const float*)d_in[5];
  const float* aWhh    = (const float*)d_in[6];
  const float* a_bih   = (const float*)d_in[7];
  const float* a_bhh   = (const float*)d_in[8];
  const float* att_Wq  = (const float*)d_in[9];
  const float* att_bq  = (const float*)d_in[10];
  const float* att_Wi  = (const float*)d_in[11];
  const float* att_bi  = (const float*)d_in[12];
  const float* att_v   = (const float*)d_in[13];
  const float* proj_W  = (const float*)d_in[14];
  const float* proj_b  = (const float*)d_in[15];
  const float* d1_Wih  = (const float*)d_in[16];
  const float* d1_Whh  = (const float*)d_in[17];
  const float* d1_bih  = (const float*)d_in[18];
  const float* d1_bhh  = (const float*)d_in[19];
  const float* d2_Wih  = (const float*)d_in[20];
  const float* d2_Whh  = (const float*)d_in[21];
  const float* d2_bih  = (const float*)d_in[22];
  const float* d2_bhh  = (const float*)d_in[23];
  const float* mel_W   = (const float*)d_in[24];
  const float* mel_b   = (const float*)d_in[25];
  const float* stop_W  = (const float*)d_in[26];
  const float* stop_b  = (const float*)d_in[27];

  char* ws = (char*)d_ws;
  size_t off = 0;
  auto take = [&](size_t bytes) -> void* {
    void* p = ws + off;
    off += (bytes + 1023) & ~(size_t)1023;
    return p;
  };

  __bf16* xbf    = (__bf16*)take((size_t)B_ * T_ * IN_ * 2);
  __bf16* procbf = (__bf16*)take((size_t)B_ * T_ * IN_ * 2);
  __bf16* pre1_p  = (__bf16*)take((size_t)16 * 3 * 1024);   // N=256 K=80->96
  __bf16* pre2_p  = (__bf16*)take((size_t)8  * 8 * 1024);   // N=128 K=256
  __bf16* aWih_p  = (__bf16*)take((size_t)48 * 4 * 1024);   // N=768 K=128
  __bf16* aWhh_p  = (__bf16*)take((size_t)48 * 8 * 1024);   // N=768 K=256
  __bf16* wq_p    = (__bf16*)take((size_t)8  * 8 * 1024);   // N=128 K=256
  __bf16* wi_p    = (__bf16*)take((size_t)8  * 4 * 1024);   // N=128 K=128
  __bf16* proj_p  = (__bf16*)take((size_t)16 * 4 * 1024);   // N=256 K=128
  __bf16* d1Wih_p = (__bf16*)take((size_t)48 * 8 * 1024);
  __bf16* d1Whh_p = (__bf16*)take((size_t)48 * 8 * 1024);
  __bf16* d2Wih_p = (__bf16*)take((size_t)48 * 8 * 1024);
  __bf16* d2Whh_p = (__bf16*)take((size_t)48 * 8 * 1024);
  __bf16* mel_p   = (__bf16*)take((size_t)25 * 8 * 1024);   // N=400 K=256

  // Phase 0: bf16 conversion + weight packing into WMMA fragment order
  cvt_bf16<<<2048, 256, 0, stream>>>(x, xbf, B_ * T_ * IN_);
  auto pack = [&](const float* W, __bf16* dst, int N, int K, int Kpad){
    int total = N * Kpad;
    pack_w<<<(total + 255) / 256, 256, 0, stream>>>(W, dst, N, K, Kpad);
  };
  pack(pre_W1, pre1_p, 256, 80, 96);
  pack(pre_W2, pre2_p, 128, 256, 256);
  pack(aWih,   aWih_p, 768, 128, 128);
  pack(aWhh,   aWhh_p, 768, 256, 256);
  pack(att_Wq, wq_p,   128, 256, 256);
  pack(att_Wi, wi_p,   128, 128, 128);
  pack(proj_W, proj_p, 256, 128, 128);
  pack(d1_Wih, d1Wih_p, 768, 256, 256);
  pack(d1_Whh, d1Whh_p, 768, 256, 256);
  pack(d2_Wih, d2Wih_p, 768, 256, 256);
  pack(d2_Whh, d2Whh_p, 768, 256, 256);
  pack(mel_W,  mel_p,  400, 256, 256);

  // Phase 1: proc_inputs GEMM (WMMA)
  proc_kernel<<<400, 256, 0, stream>>>(x, wi_p, att_bi, procbf);

  // Phase 2: persistent autoregressive decoder (4 batch tiles of 16)
  float* outp  = (float*)d_out;
  float* stops = outp + (size_t)B_ * 80 * (T_ * 5);
  decoder_kernel<<<4, 512, 0, stream>>>(
      xbf, procbf,
      pre1_p, pre_b1, pre2_p, pre_b2,
      aWih_p, aWhh_p, a_bih, a_bhh,
      wq_p, att_bq, att_v,
      proj_p, proj_b,
      d1Wih_p, d1Whh_p, d1_bih, d1_bhh,
      d2Wih_p, d2Whh_p, d2_bih, d2_bhh,
      mel_p, mel_b, stop_W, stop_b,
      outp, stops);
}
